// GCNNet_61950608278026
// MI455X (gfx1250) — compile-verified, matrix operand order checked by reading, not run on hardware
//
#include <hip/hip_runtime.h>

typedef __attribute__((ext_vector_type(16))) __bf16        v16bf;
typedef __attribute__((ext_vector_type(8)))  float         v8f;
typedef __attribute__((ext_vector_type(8)))  unsigned int  v8u;

#define F_IN  512
#define HID   16
#define NC    40
#define NCPAD 48

// ---------------- utility kernels ----------------

__global__ void zero_f32(float* __restrict__ p, long long n) {
  long long i = (long long)blockIdx.x * blockDim.x + threadIdx.x;
  if (i < n) p[i] = 0.0f;
}

__global__ void deg_kernel(const int* __restrict__ col, const float* __restrict__ w,
                           float* __restrict__ deg, int E) {
  int e = blockIdx.x * blockDim.x + threadIdx.x;
  if (e < E) atomicAdd(&deg[col[e]], w[e]);
}

__global__ void dinv_kernel(const float* __restrict__ deg, float* __restrict__ dinv, int N) {
  int i = blockIdx.x * blockDim.x + threadIdx.x;
  if (i < N) {
    float d = deg[i];
    dinv[i] = (d > 0.0f) ? rsqrtf(fmaxf(d, 1e-30f)) : 0.0f;
  }
}

__global__ void norm_kernel(const int* __restrict__ row, const int* __restrict__ col,
                            const float* __restrict__ w, const float* __restrict__ dinv,
                            float* __restrict__ nrm, int E) {
  int e = blockIdx.x * blockDim.x + threadIdx.x;
  if (e < E) nrm[e] = dinv[row[e]] * w[e] * dinv[col[e]];
}

// scatter-add of norm-scaled gathered features; F = feature count
template <int F>
__global__ void msg_kernel(const int* __restrict__ row, const int* __restrict__ col,
                           const float* __restrict__ nrm, const float* __restrict__ hsrc,
                           float* __restrict__ agg, int E) {
  long long gid = (long long)blockIdx.x * blockDim.x + threadIdx.x;
  if (gid < (long long)E * F) {
    int e = (int)(gid / F);
    int f = (int)(gid % F);
    float v = nrm[e] * hsrc[(long long)row[e] * F + f];
    atomicAdd(&agg[(long long)col[e] * F + f], v);
  }
}

__global__ void relu_bias_kernel(const float* __restrict__ agg, const float* __restrict__ b,
                                 float* __restrict__ out, long long n) {
  long long i = (long long)blockIdx.x * blockDim.x + threadIdx.x;
  if (i < n) out[i] = fmaxf(agg[i] + b[(int)(i & (HID - 1))], 0.0f);
}

// A fragment: 16x32 bf16 tile row for this lane (ISA 16-bit A layout:
// lane<16 -> khalf=0, lane>=16 -> khalf=8; elements 0..7 -> K=khalf+0..7,
// elements 8..15 -> K=khalf+16..23), converted f32 -> bf16.
__device__ __forceinline__ v16bf load_a_frag(const float* __restrict__ rowp,
                                             int kc, int khalf) {
  const float4 a0 = *(const float4*)(rowp + kc + khalf);
  const float4 a1 = *(const float4*)(rowp + kc + khalf + 4);
  const float4 a2 = *(const float4*)(rowp + kc + khalf + 16);
  const float4 a3 = *(const float4*)(rowp + kc + khalf + 20);
  v16bf a;
  a[0]  = (__bf16)a0.x; a[1]  = (__bf16)a0.y; a[2]  = (__bf16)a0.z; a[3]  = (__bf16)a0.w;
  a[4]  = (__bf16)a1.x; a[5]  = (__bf16)a1.y; a[6]  = (__bf16)a1.z; a[7]  = (__bf16)a1.w;
  a[8]  = (__bf16)a2.x; a[9]  = (__bf16)a2.y; a[10] = (__bf16)a2.z; a[11] = (__bf16)a2.w;
  a[12] = (__bf16)a3.x; a[13] = (__bf16)a3.y; a[14] = (__bf16)a3.z; a[15] = (__bf16)a3.w;
  return a;
}

// ---------------- GEMM1: hW[N,16] = x[N,512] @ W1[512,16] via bf16 WMMA ----------------
// Per wave: 32 rows (two 16-row subtiles sharing one B fragment), 32 WMMAs total.
// W1 is pre-packed in LDS in the exact B-fragment layout (bf16 pairs per u32,
// 8 contiguous u32 per lane per K-chunk) -> two ds_load_b128 per chunk.

__global__ void __launch_bounds__(256)
gemm1_wmma(const float* __restrict__ x, const float* __restrict__ W1,
           float* __restrict__ hW, int N) {
  __shared__ unsigned int w1b[(F_IN / 32) * 32 * 8];  // 16 KB
  for (int i = threadIdx.x; i < (F_IN / 32) * 32 * 8; i += blockDim.x) {
    const int v     = i & 7;
    const int ln    = (i >> 3) & 31;
    const int chunk = i >> 8;
    const int n     = ln & 15;
    const int khalf = (ln >> 4) * 8;
    const int k0    = chunk * 32 + khalf + ((v < 4) ? 2 * v : 8 + 2 * v);
    const __bf16 lo = (__bf16)W1[k0 * HID + n];
    const __bf16 hi = (__bf16)W1[(k0 + 1) * HID + n];
    w1b[i] = ((unsigned int)__builtin_bit_cast(unsigned short, hi) << 16)
           |  (unsigned int)__builtin_bit_cast(unsigned short, lo);
  }
  __syncthreads();

  const int wave = threadIdx.x >> 5;
  const int lane = threadIdx.x & 31;
  const int tile = blockIdx.x * 8 + wave;   // 32-row supertile, uniform per wave
  const int row0 = tile * 32;
  if (row0 >= N) return;                    // wave-uniform: EXEC all-ones for WMMA

  const int m     = lane & 15;
  const int half  = lane >> 4;
  const int khalf = half * 8;

  int r1 = row0 + m;        if (r1 >= N) r1 = N - 1;
  int r2 = row0 + 16 + m;   if (r2 >= N) r2 = N - 1;
  const float* xr1 = x + (long long)r1 * F_IN;
  const float* xr2 = x + (long long)r2 * F_IN;

  v8f acc0 = {}, acc1 = {};
#pragma unroll 2
  for (int kc = 0; kc < F_IN; kc += 32) {
    const unsigned int* bp = &w1b[(((kc >> 5) * 32) + lane) * 8];
    const uint4 bq0 = *(const uint4*)bp;
    const uint4 bq1 = *(const uint4*)(bp + 4);
    const v8u bu = {bq0.x, bq0.y, bq0.z, bq0.w, bq1.x, bq1.y, bq1.z, bq1.w};
    const v16bf b = __builtin_bit_cast(v16bf, bu);

    if (kc + 32 < F_IN) {
      __builtin_prefetch(xr1 + kc + 32 + khalf, 0, 3);
      __builtin_prefetch(xr2 + kc + 32 + khalf, 0, 3);
    }
    const v16bf a0 = load_a_frag(xr1, kc, khalf);
    const v16bf a1 = load_a_frag(xr2, kc, khalf);

    acc0 = __builtin_amdgcn_wmma_f32_16x16x32_bf16(false, a0, false, b,
                                                   (short)0, acc0, false, false);
    acc1 = __builtin_amdgcn_wmma_f32_16x16x32_bf16(false, a1, false, b,
                                                   (short)0, acc1, false, false);
  }

  // C/D layout: lane holds col n = lane&15, rows M = 8*half + r
  if (row0 + 32 <= N) {  // wave-uniform full-tile fast path (always taken for N%32==0)
#pragma unroll
    for (int r = 0; r < 8; ++r) {
      hW[(long long)(row0 + half * 8 + r) * HID + m]      = acc0[r];
      hW[(long long)(row0 + 16 + half * 8 + r) * HID + m] = acc1[r];
    }
  } else {
#pragma unroll
    for (int r = 0; r < 8; ++r) {
      int rr = row0 + half * 8 + r;
      if (rr < N)      hW[(long long)rr * HID + m]        = acc0[r];
      if (rr + 16 < N) hW[(long long)(rr + 16) * HID + m] = acc1[r];
    }
  }
}

// ------------- GEMM2: h2w[N,40] = h1[N,16] @ W2[16,40], K padded to 32, 3 col tiles -------------

__global__ void __launch_bounds__(256)
gemm2_wmma(const float* __restrict__ h1, const float* __restrict__ W2,
           float* __restrict__ h2w, int N) {
  __shared__ float w2s[HID * NCPAD];  // padded to 48 cols
  for (int i = threadIdx.x; i < HID * NCPAD; i += blockDim.x) {
    const int k = i / NCPAD, c = i % NCPAD;
    w2s[i] = (c < NC) ? W2[k * NC + c] : 0.0f;
  }
  __syncthreads();

  const int wave = threadIdx.x >> 5;
  const int lane = threadIdx.x & 31;
  const int rowTiles = (N + 15) >> 4;
  const int t = blockIdx.x * 8 + wave;      // uniform per wave
  if (t >= rowTiles * 3) return;
  const int rowtile = t / 3;
  const int ct      = t % 3;
  const int row0 = rowtile * 16;
  const int col0 = ct * 16;

  const int m     = lane & 15;
  const int half  = lane >> 4;
  const int khalf = half * 8;

  int arow = row0 + m;
  if (arow >= N) arow = N - 1;
  const float* hrow = h1 + (long long)arow * HID;

  v16bf a, b;
#pragma unroll
  for (int e = 0; e < 16; ++e) {
    const int k = khalf + ((e < 8) ? e : (16 + e - 8));   // 0..31, pad K>=16 with 0
    a[e] = (k < HID) ? (__bf16)hrow[k] : (__bf16)0.0f;
    b[e] = (k < HID) ? (__bf16)w2s[k * NCPAD + col0 + m] : (__bf16)0.0f;
  }

  v8f acc = {};
  acc = __builtin_amdgcn_wmma_f32_16x16x32_bf16(false, a, false, b,
                                                (short)0, acc, false, false);

  const int cc = col0 + m;
  if (row0 + 16 <= N) {   // wave-uniform full-tile fast path
    if (cc < NC) {
#pragma unroll
      for (int r = 0; r < 8; ++r)
        h2w[(long long)(row0 + half * 8 + r) * NC + cc] = acc[r];
    }
  } else {
#pragma unroll
    for (int r = 0; r < 8; ++r) {
      const int rr = row0 + half * 8 + r;
      if (rr < N && cc < NC) h2w[(long long)rr * NC + cc] = acc[r];
    }
  }
}

// ---------------- log_softmax over 40 classes, one wave32 per node ----------------

__global__ void lsm_kernel(const float* __restrict__ agg2, const float* __restrict__ b2,
                           float* __restrict__ out, int N) {
  const int wid  = (int)(((long long)blockIdx.x * blockDim.x + threadIdx.x) >> 5);
  const int lane = threadIdx.x & 31;
  if (wid >= N) return;
  const float* rowp = agg2 + (long long)wid * NC;

  const bool hi = (lane < NC - 32);
  float v0 = rowp[lane] + b2[lane];
  float v1 = hi ? (rowp[32 + lane] + b2[32 + lane]) : -3.0e38f;

  float mx = fmaxf(v0, v1);
#pragma unroll
  for (int off = 16; off > 0; off >>= 1) mx = fmaxf(mx, __shfl_xor(mx, off, 32));

  float s = __expf(v0 - mx) + (hi ? __expf(v1 - mx) : 0.0f);
#pragma unroll
  for (int off = 16; off > 0; off >>= 1) s += __shfl_xor(s, off, 32);

  const float lse = mx + __logf(s);
  out[(long long)wid * NC + lane] = v0 - lse;
  if (hi) out[(long long)wid * NC + 32 + lane] = v1 - lse;
}

// ---------------- launcher ----------------

extern "C" void kernel_launch(void* const* d_in, const int* in_sizes, int n_in,
                              void* d_out, int out_size, void* d_ws, size_t ws_size,
                              hipStream_t stream) {
  const float* x  = (const float*)d_in[0];
  const int*   ei = (const int*)d_in[1];
  const float* ew = (const float*)d_in[2];
  const float* W1 = (const float*)d_in[3];
  const float* b1 = (const float*)d_in[4];
  const float* W2 = (const float*)d_in[5];
  const float* b2 = (const float*)d_in[6];

  const int N = in_sizes[0] / F_IN;
  const int E = in_sizes[2];
  const int* row = ei;
  const int* col = ei + E;

  float* ws   = (float*)d_ws;
  float* deg  = ws;
  float* dinv = deg  + N;
  float* nrm  = dinv + N;
  float* hW   = nrm  + E;                       // x@W1, later reused as h1
  float* agg1 = hW   + (long long)N * HID;
  float* h2w  = agg1 + (long long)N * HID;
  float* agg2 = h2w  + (long long)N * NC;
  float* outp = (float*)d_out;

  const int B = 256;
  auto blocks = [](long long n, int b) { return (unsigned)((n + b - 1) / b); };

  // zero-init accumulators (ws is poisoned by the harness)
  zero_f32<<<blocks(N, B), B, 0, stream>>>(deg, N);
  zero_f32<<<blocks((long long)N * HID, B), B, 0, stream>>>(agg1, (long long)N * HID);
  zero_f32<<<blocks((long long)N * NC, B), B, 0, stream>>>(agg2, (long long)N * NC);

  // symmetric GCN norm
  deg_kernel <<<blocks(E, B), B, 0, stream>>>(col, ew, deg, E);
  dinv_kernel<<<blocks(N, B), B, 0, stream>>>(deg, dinv, N);
  norm_kernel<<<blocks(E, B), B, 0, stream>>>(row, col, ew, dinv, nrm, E);

  // layer 1: GEMM (WMMA) -> scatter -> bias+relu
  {
    const int tiles = (N + 31) / 32;
    gemm1_wmma<<<blocks(tiles, 8), B, 0, stream>>>(x, W1, hW, N);
  }
  msg_kernel<HID><<<blocks((long long)E * HID, B), B, 0, stream>>>(row, col, nrm, hW, agg1, E);
  relu_bias_kernel<<<blocks((long long)N * HID, B), B, 0, stream>>>(agg1, b1, hW, (long long)N * HID);

  // layer 2: GEMM (WMMA, K padded) -> scatter -> bias + log_softmax
  {
    const int tiles = ((N + 15) / 16) * 3;
    gemm2_wmma<<<blocks(tiles, 8), B, 0, stream>>>(hW, W2, h2w, N);
  }
  msg_kernel<NC><<<blocks((long long)E * NC, B), B, 0, stream>>>(row, col, nrm, h2w, agg2, E);
  lsm_kernel<<<blocks((long long)N * 32, B), B, 0, stream>>>(agg2, b2, outp, N);
}
